// EdgeLayer_37074157699486
// MI455X (gfx1250) — matrix operand order repaired and checked by reading, not compiled, not run on hardware
//
#include <hip/hip_runtime.h>
#include <hip/hip_bf16.h>
#include <stdint.h>

#define D_IN     32
#define D_HID    128
#define D_NODE   16
#define MT       64      // edges per tile
#define STRIDE_A 40      // bf16/row, staged edge features (32 + 8 pad)
#define STRIDE_H 136     // bf16/row, hidden activations (128 + 8 pad)
#define STRIDE_N 20      // f32/row, staged node features (16 + 4 pad, keeps 16B align)
#define STRIDE_O 17      // f32/row, staged outputs (16 + 1 pad)

typedef __attribute__((ext_vector_type(16))) __bf16 v16bf;
typedef __attribute__((ext_vector_type(8)))  float  v8f;

struct __attribute__((aligned(16))) U4 { uint32_t x, y, z, w; };
struct __attribute__((aligned(16))) F4 { float x, y, z, w; };

union Frag { v16bf v; uint32_t u[8]; U4 q[2]; };

// Packed f32->bf16 (RNE) via native instruction (builtin not declared for gfx1250).
__device__ __forceinline__ uint32_t pack_bf(float lo, float hi) {
  uint32_t r;
  asm("v_cvt_pk_bf16_f32 %0, %1, %2" : "=v"(r) : "v"(lo), "v"(hi));
  return r;
}

// Pack two f32 to bf16, then packed-ReLU in one VOP3P op (avoids the
// canonicalize+max pair LLVM emits for IEEE fmax on f32).
__device__ __forceinline__ uint32_t pack_bf_relu(float lo, float hi) {
  uint32_t r;
  asm("v_cvt_pk_bf16_f32 %0, %1, %2" : "=v"(r) : "v"(lo), "v"(hi));
  asm("v_pk_max_num_bf16 %0, %1, 0" : "=v"(r) : "v"(r));
  return r;
}

// B-fragment (KxN = 32x16) for W row-major [n_out][k_in]: B[k][c] = W[(16n+c)*kin + kb*32 + k]
__device__ __forceinline__ void load_bfrag(Frag& f, const float* __restrict__ W,
                                           int kin, int ntile, int kb, int lane) {
  const int c    = lane & 15;
  const int kOff = (lane & 16) ? 8 : 0;
  const float* p = W + (size_t)(16 * ntile + c) * kin + kb * 32 + kOff;
#pragma unroll
  for (int v = 0; v < 4; ++v) f.u[v]     = pack_bf(p[2 * v],      p[2 * v + 1]);
#pragma unroll
  for (int v = 0; v < 4; ++v) f.u[4 + v] = pack_bf(p[16 + 2 * v], p[16 + 2 * v + 1]);
}

// A-fragment (MxK = 16x32) for W row-major: A row = W row (16n + (lane&15)).
__device__ __forceinline__ void load_afrag_w(Frag& f, const float* __restrict__ W,
                                             int kin, int ntile, int kb, int lane) {
  const int r    = lane & 15;
  const int kSel = (lane & 16) ? 8 : 0;
  const float* p = W + (size_t)(16 * ntile + r) * kin + kb * 32 + kSel;
#pragma unroll
  for (int v = 0; v < 4; ++v) f.u[v]     = pack_bf(p[2 * v],      p[2 * v + 1]);
#pragma unroll
  for (int v = 0; v < 4; ++v) f.u[4 + v] = pack_bf(p[16 + 2 * v], p[16 + 2 * v + 1]);
}

// A-fragment from row-major bf16 LDS; rowptr = row (lane&15) base + k-block.
__device__ __forceinline__ void load_afrag(Frag& f, const __bf16* rowptr, int lane) {
  const int kSel = (lane & 16) ? 8 : 0;
  f.q[0] = *(const U4*)(rowptr + kSel);
  f.q[1] = *(const U4*)(rowptr + kSel + 16);
}

// B-fragment of Hᵀ from row-major bf16 LDS (lane column e = lane&15).
__device__ __forceinline__ void load_bfrag_lds(Frag& f, const __bf16* rowptr, int lane) {
  const int kOff = (lane & 16) ? 8 : 0;
  f.q[0] = *(const U4*)(rowptr + kOff);
  f.q[1] = *(const U4*)(rowptr + kOff + 16);
}

__device__ __forceinline__ v8f wmma_bf16(const Frag& a, const Frag& b, v8f c) {
  return __builtin_amdgcn_wmma_f32_16x16x32_bf16(false, a.v, false, b.v,
                                                 (short)0, c, false, false);
}

// relu(acc) -> bf16 row-store into H (8 rows of one column per lane).
__device__ __forceinline__ void store_h(__bf16* sH, int rowb, int cc, const v8f& acc) {
#pragma unroll
  for (int r = 0; r < 8; r += 2) {
    uint32_t pk = pack_bf_relu(acc[r], acc[r + 1]);
    *(uint16_t*)(sH + (rowb + r    ) * STRIDE_H + cc) = (uint16_t)pk;
    *(uint16_t*)(sH + (rowb + r + 1) * STRIDE_H + cc) = (uint16_t)(pk >> 16);
  }
}

// 128-K layer for one 16-edge subtile: preload 4 k-fragments, chain 4 WMMAs.
__device__ __forceinline__ v8f layer_k128(const __bf16* rp, const Frag* Bf, v8f bias, int lane) {
  Frag Ak[4];
#pragma unroll
  for (int kb = 0; kb < 4; ++kb) load_afrag(Ak[kb], rp + kb * 32, lane);
  v8f acc = wmma_bf16(Ak[0], Bf[0], bias);
#pragma unroll
  for (int kb = 1; kb < 4; ++kb) acc = wmma_bf16(Ak[kb], Bf[kb], acc);
  return acc;
}

__device__ __forceinline__ void gather_tile(const float* __restrict__ ef,
                                            const float* __restrict__ nfeat,
                                            const int* __restrict__ eidx,
                                            __bf16* sA, float* sNF,
                                            int eBase, int E, int tid) {
  const int el = tid >> 2;
  const int p  = tid & 3;
  int e = eBase + el;
  if (e >= E) e = E - 1;
  const int nb = eidx[2 * e + 1];
  const float* efp = ef + (size_t)nb * D_IN + p * 8;
  U4 d;
  d.x = pack_bf(efp[0], efp[1]);
  d.y = pack_bf(efp[2], efp[3]);
  d.z = pack_bf(efp[4], efp[5]);
  d.w = pack_bf(efp[6], efp[7]);
  *(U4*)(sA + el * STRIDE_A + p * 8) = d;
  *(F4*)(sNF + el * STRIDE_N + p * 4) = *(const F4*)(nfeat + (size_t)nb * D_NODE + p * 4);
}

__global__ __launch_bounds__(256)
void edge_mlp_wmma(const float* __restrict__ ef, const float* __restrict__ nfeat,
                   const float* __restrict__ W1, const float* __restrict__ b1,
                   const float* __restrict__ W2, const float* __restrict__ b2,
                   const float* __restrict__ W3, const float* __restrict__ b3,
                   const float* __restrict__ W4, const float* __restrict__ b4,
                   const int* __restrict__ eidx, float* __restrict__ out, int E)
{
  __shared__ __bf16 sA  [MT * STRIDE_A];
  __shared__ __bf16 sH1 [MT * STRIDE_H];
  __shared__ __bf16 sH2 [MT * STRIDE_H];
  __shared__ float  sNF [MT * STRIDE_N];
  __shared__ float  sOUT[MT * STRIDE_O];

  const int tid  = threadIdx.x;
  const int lane = tid & 31;
  const int wv   = tid >> 5;          // wave id == N-tile for layers 1-3
  const int col  = lane & 15;
  const int hi   = (lane >> 4) & 1;

  // ---- register-resident bf16 weight fragments ----
  Frag B1f;            load_bfrag(B1f, W1, D_IN, wv, 0, lane);
  Frag B2f[4], B3f[4], A4f[2][4];
#pragma unroll
  for (int kb = 0; kb < 4; ++kb) {
    load_bfrag  (B2f[kb],    W2, D_HID, wv,         kb, lane);
    load_bfrag  (B3f[kb],    W3, D_HID, wv,         kb, lane);
    load_afrag_w(A4f[0][kb], W4, D_HID, 2 * wv,     kb, lane);
    load_afrag_w(A4f[1][kb], W4, D_HID, 2 * wv + 1, kb, lane);
  }
  // ---- bias vectors, fed directly as WMMA C operand ----
  v8f bv1, bv2, bv3, bv4[2];
  {
    const float x1 = b1[16 * wv + col];
    const float x2 = b2[16 * wv + col];
    const float x3 = b3[16 * wv + col];
#pragma unroll
    for (int i = 0; i < 8; ++i) { bv1[i] = x1; bv2[i] = x2; bv3[i] = x3; }
#pragma unroll
    for (int h = 0; h < 2; ++h)
#pragma unroll
      for (int r = 0; r < 8; ++r)
        bv4[h][r] = b4[16 * (2 * wv + h) + 8 * hi + r];   // bias by output row
  }

  const int numTiles = (E + MT - 1) / MT;
  const int firstTile = blockIdx.x;
  if (firstTile < numTiles)
    gather_tile(ef, nfeat, eidx, sA, sNF, firstTile * MT, E, tid);

  for (int tile = firstTile; tile < numTiles; tile += gridDim.x) {
    const int eBase = tile * MT;
    __syncthreads();   // gather visible

    // ---- Layer 1: sA (64x32) -> relu -> sH1 (all 4 A-frags preloaded) ----
    {
      Frag Aa[4];
#pragma unroll
      for (int ms = 0; ms < 4; ++ms)
        load_afrag(Aa[ms], sA + (ms * 16 + col) * STRIDE_A, lane);
#pragma unroll
      for (int ms = 0; ms < 4; ++ms) {
        v8f acc = wmma_bf16(Aa[ms], B1f, bv1);
        store_h(sH1, ms * 16 + 8 * hi, 16 * wv + col, acc);
      }
    }
    __syncthreads();

    // ---- Layer 2: sH1 -> relu -> sH2 ----
#pragma unroll
    for (int ms = 0; ms < 4; ++ms) {
      v8f acc = layer_k128(sH1 + (ms * 16 + col) * STRIDE_H, B2f, bv2, lane);
      store_h(sH2, ms * 16 + 8 * hi, 16 * wv + col, acc);
    }
    __syncthreads();

    // ---- Layer 3: sH2 -> relu -> sH1 ----
#pragma unroll
    for (int ms = 0; ms < 4; ++ms) {
      v8f acc = layer_k128(sH2 + (ms * 16 + col) * STRIDE_H, B3f, bv3, lane);
      store_h(sH1, ms * 16 + 8 * hi, 16 * wv + col, acc);
    }
    __syncthreads();

    // ---- Layer 4 transposed: D' = W4tile (A) x H3ᵀ (B); o in register dim.
    // lane holds EO[e=col][i=2w+h][j=8*hi+r] in acc[r]; einsum = in-lane dot.
#pragma unroll
    for (int ms = 0; ms < 4; ++ms) {
      const __bf16* rp = sH1 + (ms * 16 + col) * STRIDE_H;
      Frag Bt[4];
#pragma unroll
      for (int kb = 0; kb < 4; ++kb) load_bfrag_lds(Bt[kb], rp + kb * 32, lane);

      const float* np = sNF + (ms * 16 + col) * STRIDE_N + 8 * hi;
      F4 n0 = *(const F4*)np;
      F4 n1 = *(const F4*)(np + 4);
      const float nfv[8] = {n0.x, n0.y, n0.z, n0.w, n1.x, n1.y, n1.z, n1.w};

#pragma unroll
      for (int h = 0; h < 2; ++h) {
        v8f acc = wmma_bf16(A4f[h][0], Bt[0], bv4[h]);
#pragma unroll
        for (int kb = 1; kb < 4; ++kb)
          acc = wmma_bf16(A4f[h][kb], Bt[kb], acc);
        float dot = 0.0f;
#pragma unroll
        for (int r = 0; r < 8; ++r) dot = fmaf(acc[r], nfv[r], dot);
        dot += __shfl_xor(dot, 16);          // combine j=0..7 with j=8..15
        if (lane < 16)
          sOUT[(ms * 16 + col) * STRIDE_O + 2 * wv + h] = dot;
      }
    }
    __syncthreads();   // sOUT ready; sA/sNF free

    // ---- coalesced output copy + next tile's gather (disjoint LDS buffers) ----
    {
      const int el = tid >> 2;
      const int p  = tid & 3;
      const int e  = eBase + el;
      if (e < E) {
        F4 o;
        o.x = sOUT[el * STRIDE_O + p * 4 + 0];
        o.y = sOUT[el * STRIDE_O + p * 4 + 1];
        o.z = sOUT[el * STRIDE_O + p * 4 + 2];
        o.w = sOUT[el * STRIDE_O + p * 4 + 3];
        *(F4*)(out + (size_t)e * D_NODE + p * 4) = o;
      }
    }
    const int nextTile = tile + gridDim.x;
    if (nextTile < numTiles)
      gather_tile(ef, nfeat, eidx, sA, sNF, nextTile * MT, E, tid);
  }
}

extern "C" void kernel_launch(void* const* d_in, const int* in_sizes, int n_in,
                              void* d_out, int out_size, void* d_ws, size_t ws_size,
                              hipStream_t stream) {
  const float* ef = (const float*)d_in[0];
  const float* nf = (const float*)d_in[1];
  const float* W1 = (const float*)d_in[2];
  const float* b1 = (const float*)d_in[3];
  const float* W2 = (const float*)d_in[4];
  const float* b2 = (const float*)d_in[5];
  const float* W3 = (const float*)d_in[6];
  const float* b3 = (const float*)d_in[7];
  const float* W4 = (const float*)d_in[8];
  const float* b4 = (const float*)d_in[9];
  const int* eidx = (const int*)d_in[10];
  float* out = (float*)d_out;

  const int E = in_sizes[10] / 2;
  const int numTiles = (E + MT - 1) / MT;
  const int grid = numTiles < 1024 ? numTiles : 1024;
  edge_mlp_wmma<<<grid, 256, 0, stream>>>(ef, nf, W1, b1, W2, b2, W3, b3, W4, b4,
                                          eidx, out, E);
}
